// SelfAttention_35622458753622
// MI455X (gfx1250) — compile-verified, matrix operand order checked by reading, not run on hardware
//
#include <hip/hip_runtime.h>

#define DIM    1024
#define HEADS  16
#define HD     64
#define SEQ    1024
#define BATCH  8
#define NTOK   (BATCH * SEQ)   // 8192
#define QKV_N  (3 * DIM)       // 3072

typedef __attribute__((ext_vector_type(16))) __bf16 v16bf;
typedef __attribute__((ext_vector_type(8)))  __bf16 v8bf;
typedef __attribute__((ext_vector_type(8)))  float  v8f;

// Load one 16-bf16 WMMA A/B fragment for this lane from an LDS row:
// per CDNA5 ISA 16-bit layout, lane holds k = base+0..7 and k = base+16..23
// with base = (lane>>4)*8, row = lane&15. k0 = chunk*32 + (lane>>4)*8.
__device__ __forceinline__ v16bf ld_frag(const __bf16* row, int k0) {
    v8bf lo = *reinterpret_cast<const v8bf*>(row + k0);
    v8bf hi = *reinterpret_cast<const v8bf*>(row + k0 + 16);
    return __builtin_shufflevector(lo, hi, 0,1,2,3,4,5,6,7,8,9,10,11,12,13,14,15);
}

__device__ __forceinline__ v8f wmma_bf16(v16bf a, v16bf b, v8f c) {
    return __builtin_amdgcn_wmma_f32_16x16x32_bf16(false, a, false, b, (short)0, c, false, false);
}

// Async copy 16B global -> LDS (GLOBAL_LOAD_ASYNC_TO_LDS_B128, ASYNCcnt).
// Generic LDS pointer: low 32 bits are the LDS byte offset (flat aperture rule).
__device__ __forceinline__ void async_copy16(void* lds_dst, const void* gsrc) {
    unsigned loff = (unsigned)(uintptr_t)lds_dst;
    asm volatile("global_load_async_to_lds_b128 %0, %1, off"
                 :: "v"(loff), "v"(gsrc) : "memory");
}
__device__ __forceinline__ void wait_async0() {
#if __has_builtin(__builtin_amdgcn_s_wait_asynccnt)
    __builtin_amdgcn_s_wait_asynccnt(0);
#else
    asm volatile("s_wait_asynccnt 0x0" ::: "memory");
#endif
}

// ---------------------------------------------------------------- K0: f32 -> bf16
__global__ void cvt_bf16_kernel(const float* __restrict__ src,
                                __bf16* __restrict__ dst, int n) {
    int i = (blockIdx.x * blockDim.x + threadIdx.x) * 4;
    if (i < n) {
        float4 v = *reinterpret_cast<const float4*>(src + i);
        dst[i + 0] = (__bf16)v.x;
        dst[i + 1] = (__bf16)v.y;
        dst[i + 2] = (__bf16)v.z;
        dst[i + 3] = (__bf16)v.w;
    }
}

// ---------------------------------------------------------------- K1: RMSNorm
__global__ __launch_bounds__(256) void rmsnorm_kernel(const float* __restrict__ x,
                                                      const float* __restrict__ scale,
                                                      __bf16* __restrict__ xn) {
    __shared__ float red[8];
    int tok = blockIdx.x;
    int tid = threadIdx.x;
    float4 v = reinterpret_cast<const float4*>(x + (size_t)tok * DIM)[tid];
    float ss = v.x * v.x + v.y * v.y + v.z * v.z + v.w * v.w;
    #pragma unroll
    for (int off = 16; off > 0; off >>= 1) ss += __shfl_xor(ss, off, 32);
    if ((tid & 31) == 0) red[tid >> 5] = ss;
    __syncthreads();
    float tot = 0.f;
    #pragma unroll
    for (int i = 0; i < 8; i++) tot += red[i];
    float s = rsqrtf(tot * (1.0f / DIM) + 1e-6f);
    float4 sc = reinterpret_cast<const float4*>(scale)[tid];
    __bf16* out = xn + (size_t)tok * DIM + tid * 4;
    out[0] = (__bf16)(v.x * sc.x * s);
    out[1] = (__bf16)(v.y * sc.y * s);
    out[2] = (__bf16)(v.z * sc.z * s);
    out[3] = (__bf16)(v.w * sc.w * s);
}

// ---------------------------------------------------------------- K2: QKV GEMM + RoPE
// C[8192,3072] = Xn[8192,1024] * Wqkv[3072,1024]^T  (NT: both K-major)
// Double-buffered LDS, async-to-LDS copies, K-step 64, one barrier/step.
__global__ __launch_bounds__(256) void qkv_gemm_rope_kernel(
        const __bf16* __restrict__ xn, const __bf16* __restrict__ wqkv,
        const float* __restrict__ theta,
        __bf16* __restrict__ qd, __bf16* __restrict__ kd, __bf16* __restrict__ vd) {
    __shared__ __align__(16) __bf16 As[2][128][72];
    __shared__ __align__(16) __bf16 Bs[2][64][72];
    const int n0 = blockIdx.x * 64;    // 48 tiles over 3072
    const int m0 = blockIdx.y * 128;   // 64 tiles over 8192
    const int tid = threadIdx.x;
    const int wave = tid >> 5, lane = tid & 31;
    const int ln = lane & 15, hi = lane >> 4;

    const int ar = tid >> 1, ac = (tid & 1) * 32;   // A: 128 x 64, 64B/thread
    const int br = tid >> 2, bc = (tid & 3) * 16;   // B:  64 x 64, 32B/thread
    const __bf16* aB = xn   + (size_t)(m0 + ar) * DIM + ac;
    const __bf16* bB = wqkv + (size_t)(n0 + br) * DIM + bc;

    auto load_tile = [&](int k0, int buf) {
        #pragma unroll
        for (int u = 0; u < 4; u++)
            async_copy16(&As[buf][ar][ac + u * 8], aB + k0 + u * 8);
        #pragma unroll
        for (int u = 0; u < 2; u++)
            async_copy16(&Bs[buf][br][bc + u * 8], bB + k0 + u * 8);
    };

    load_tile(0, 0);
    wait_async0();
    __syncthreads();

    v8f acc[4] = {};
    const int NSTEP = DIM / 64;   // 16
    for (int i = 0; i < NSTEP; i++) {
        int cur = i & 1;
        if (i + 1 < NSTEP) load_tile((i + 1) * 64, cur ^ 1);
        #pragma unroll
        for (int kk = 0; kk < 2; kk++) {
            v16bf a = ld_frag(&As[cur][wave * 16 + ln][0], kk * 32 + hi * 8);
            #pragma unroll
            for (int nf = 0; nf < 4; nf++) {
                v16bf b = ld_frag(&Bs[cur][nf * 16 + ln][0], kk * 32 + hi * 8);
                acc[nf] = wmma_bf16(a, b, acc[nf]);
            }
        }
        if (i + 1 < NSTEP) wait_async0();
        __syncthreads();
    }

    const int comp = n0 >> 10;            // 0=q 1=k 2=v
    const int h = (n0 & 1023) >> 6;       // head (tile never crosses a head)
    #pragma unroll
    for (int r = 0; r < 8; r++) {
        int tok = m0 + wave * 16 + r + 8 * hi;
        int b = tok >> 10, seq = tok & 1023;
        float v0 = acc[0][r], v1 = acc[1][r], v2 = acc[2][r], v3 = acc[3][r];
        if (comp == 2) {
            // V stored transposed: [B,H,D,S]
            __bf16* vp = vd + ((size_t)(b * HEADS + h) * HD) * SEQ + seq;
            vp[(0 * 16 + ln) * SEQ] = (__bf16)v0;
            vp[(1 * 16 + ln) * SEQ] = (__bf16)v1;
            vp[(2 * 16 + ln) * SEQ] = (__bf16)v2;
            vp[(3 * 16 + ln) * SEQ] = (__bf16)v3;
        } else {
            // RoPE on d<32: pair (d, d+16) = (frag0, frag1), same lane slot
            float th = theta[seq * (HEADS * 16) + h * 16 + ln];
            float sn, cs;
            __sincosf(th, &sn, &cs);
            float y0 = v0 * cs - v1 * sn;
            float y1 = v1 * cs + v0 * sn;
            __bf16* dst = (comp == 0 ? qd : kd) + ((size_t)(b * HEADS + h) * SEQ + seq) * HD;
            dst[0 * 16 + ln] = (__bf16)y0;
            dst[1 * 16 + ln] = (__bf16)y1;
            dst[2 * 16 + ln] = (__bf16)v2;
            dst[3 * 16 + ln] = (__bf16)v3;
        }
    }
}

// ---------------------------------------------------------------- K3: causal flash attention
__global__ __launch_bounds__(256) void attention_kernel(
        const __bf16* __restrict__ qd, const __bf16* __restrict__ kd,
        const __bf16* __restrict__ vd, __bf16* __restrict__ od) {
    __shared__ __align__(16) __bf16 Ks[2][64][72];
    __shared__ __align__(16) __bf16 Vs[2][64][72];
    __shared__ __align__(16) __bf16 Ps[8][16][72];
    const int qt = blockIdx.x & 7;
    const int bh = blockIdx.x >> 3;   // b*16+h
    const int b  = bh >> 4, h = bh & 15;
    const int q0 = qt * 128;
    const int tid = threadIdx.x, wave = tid >> 5, lane = tid & 31;
    const int ln = lane & 15, hi = lane >> 4;

    // Q fragments resident in registers for the whole tile
    const __bf16* qrow = qd + ((size_t)bh * SEQ + q0 + wave * 16 + ln) * HD;
    v16bf qfrag[2];
    qfrag[0] = ld_frag(qrow, 0 * 32 + hi * 8);
    qfrag[1] = ld_frag(qrow, 1 * 32 + hi * 8);

    const int kr = tid >> 2, kc = (tid & 3) * 16;   // 64 x 64 chunk, 32B/thread
    const __bf16* kBp = kd + ((size_t)bh * SEQ + kr) * HD + kc;   // + j0*HD
    const __bf16* vBp = vd + ((size_t)bh * HD + kr) * SEQ + kc;   // + j0
    auto load_chunk = [&](int j0, int buf) {
        #pragma unroll
        for (int u = 0; u < 2; u++)
            async_copy16(&Ks[buf][kr][kc + u * 8], kBp + (size_t)j0 * HD + u * 8);
        #pragma unroll
        for (int u = 0; u < 2; u++)
            async_copy16(&Vs[buf][kr][kc + u * 8], vBp + j0 + u * 8);
    };

    load_chunk(0, 0);
    wait_async0();
    __syncthreads();

    v8f accO[4] = {};
    float mrun[8], lrun[8];
    #pragma unroll
    for (int r = 0; r < 8; r++) { mrun[r] = -1e30f; lrun[r] = 0.f; }

    const int nchunks = (q0 + 128) / 64;
    for (int ci = 0; ci < nchunks; ci++) {
        const int j0 = ci * 64, cur = ci & 1;
        if (ci + 1 < nchunks) load_chunk(j0 + 64, cur ^ 1);

        // scores: Q(16x64d) * K(64keys x 64d)^T  -> 4 frags of 16x16
        v8f accS[4] = {};
        #pragma unroll
        for (int nf = 0; nf < 4; nf++) {
            #pragma unroll
            for (int kk = 0; kk < 2; kk++) {
                v16bf bf = ld_frag(&Ks[cur][nf * 16 + ln][0], kk * 32 + hi * 8);
                accS[nf] = wmma_bf16(qfrag[kk], bf, accS[nf]);
            }
        }

        // online softmax per row (rows of a C-frag live in one 16-lane half)
        #pragma unroll
        for (int r = 0; r < 8; r++) {
            int qg = q0 + wave * 16 + r + 8 * hi;
            float sv[4];
            #pragma unroll
            for (int nf = 0; nf < 4; nf++) {
                int kg = j0 + nf * 16 + ln;
                float v = accS[nf][r] * 0.125f;   // 1/sqrt(64)
                sv[nf] = (kg <= qg) ? v : -1e30f;
            }
            float rmax = fmaxf(fmaxf(sv[0], sv[1]), fmaxf(sv[2], sv[3]));
            #pragma unroll
            for (int off = 8; off > 0; off >>= 1) rmax = fmaxf(rmax, __shfl_xor(rmax, off, 16));
            float mnew = fmaxf(mrun[r], rmax);
            float corr = __expf(mrun[r] - mnew);
            float psum = 0.f;
            #pragma unroll
            for (int nf = 0; nf < 4; nf++) {
                float p = __expf(sv[nf] - mnew);
                psum += p;
                Ps[wave][r + 8 * hi][nf * 16 + ln] = (__bf16)p;
            }
            #pragma unroll
            for (int off = 8; off > 0; off >>= 1) psum += __shfl_xor(psum, off, 16);
            lrun[r] = lrun[r] * corr + psum;
            mrun[r] = mnew;
            #pragma unroll
            for (int df = 0; df < 4; df++) accO[df][r] = accO[df][r] * corr;
        }
        // same-wave LDS round trip: enforce order + block compiler reordering
        asm volatile("s_wait_dscnt 0x0" ::: "memory");

        // O += P(16x64keys) * V(64keys x 64d)
        #pragma unroll
        for (int df = 0; df < 4; df++) {
            #pragma unroll
            for (int kk = 0; kk < 2; kk++) {
                v16bf pf = ld_frag(&Ps[wave][ln][0], kk * 32 + hi * 8);
                v16bf vf = ld_frag(&Vs[cur][df * 16 + ln][0], kk * 32 + hi * 8);
                accO[df] = wmma_bf16(pf, vf, accO[df]);
            }
        }
        if (ci + 1 < nchunks) wait_async0();
        __syncthreads();
    }

    #pragma unroll
    for (int r = 0; r < 8; r++) {
        int sg = q0 + wave * 16 + r + 8 * hi;
        float inv = 1.0f / lrun[r];
        __bf16* dst = od + ((size_t)(b * SEQ + sg)) * DIM + h * HD;
        #pragma unroll
        for (int df = 0; df < 4; df++) dst[df * 16 + ln] = (__bf16)(accO[df][r] * inv);
    }
}

// ---------------------------------------------------------------- K4: out-proj + residual
// out[8192,1024] = O[8192,1024] * Wout[1024,1024]^T + x
__global__ __launch_bounds__(256) void outproj_kernel(
        const __bf16* __restrict__ o, const __bf16* __restrict__ wout,
        const float* __restrict__ x, float* __restrict__ out) {
    __shared__ __align__(16) __bf16 As[2][128][72];
    __shared__ __align__(16) __bf16 Bs[2][64][72];
    const int n0 = blockIdx.x * 64;
    const int m0 = blockIdx.y * 128;
    const int tid = threadIdx.x;
    const int wave = tid >> 5, lane = tid & 31;
    const int ln = lane & 15, hi = lane >> 4;

    const int ar = tid >> 1, ac = (tid & 1) * 32;
    const int br = tid >> 2, bc = (tid & 3) * 16;
    const __bf16* aB = o    + (size_t)(m0 + ar) * DIM + ac;
    const __bf16* bB = wout + (size_t)(n0 + br) * DIM + bc;

    auto load_tile = [&](int k0, int buf) {
        #pragma unroll
        for (int u = 0; u < 4; u++)
            async_copy16(&As[buf][ar][ac + u * 8], aB + k0 + u * 8);
        #pragma unroll
        for (int u = 0; u < 2; u++)
            async_copy16(&Bs[buf][br][bc + u * 8], bB + k0 + u * 8);
    };

    load_tile(0, 0);
    wait_async0();
    __syncthreads();

    v8f acc[4] = {};
    const int NSTEP = DIM / 64;
    for (int i = 0; i < NSTEP; i++) {
        int cur = i & 1;
        if (i + 1 < NSTEP) load_tile((i + 1) * 64, cur ^ 1);
        #pragma unroll
        for (int kk = 0; kk < 2; kk++) {
            v16bf a = ld_frag(&As[cur][wave * 16 + ln][0], kk * 32 + hi * 8);
            #pragma unroll
            for (int nf = 0; nf < 4; nf++) {
                v16bf b = ld_frag(&Bs[cur][nf * 16 + ln][0], kk * 32 + hi * 8);
                acc[nf] = wmma_bf16(a, b, acc[nf]);
            }
        }
        if (i + 1 < NSTEP) wait_async0();
        __syncthreads();
    }

    #pragma unroll
    for (int r = 0; r < 8; r++) {
        int tok = m0 + wave * 16 + r + 8 * hi;
        float* dst        = out + (size_t)tok * DIM + n0;
        const float* skip = x   + (size_t)tok * DIM + n0;
        #pragma unroll
        for (int nf = 0; nf < 4; nf++)
            dst[nf * 16 + ln] = acc[nf][r] + skip[nf * 16 + ln];
    }
}

// ---------------------------------------------------------------- launch
extern "C" void kernel_launch(void* const* d_in, const int* in_sizes, int n_in,
                              void* d_out, int out_size, void* d_ws, size_t ws_size,
                              hipStream_t stream) {
    const float* x     = (const float*)d_in[0];   // [8,1024,1024]
    const float* scale = (const float*)d_in[1];   // [1024]
    const float* wqkv  = (const float*)d_in[2];   // [3072,1024]
    const float* wout  = (const float*)d_in[3];   // [1024,1024]
    const float* theta = (const float*)d_in[4];   // [1024,16,16]

    char* ws = (char*)d_ws;
    __bf16* wqkv_bf = (__bf16*)(ws);                         //  6,291,456 B
    __bf16* wout_bf = (__bf16*)(ws + 6291456);               //  2,097,152 B
    __bf16* xn_bf   = (__bf16*)(ws + 8388608);               // 16,777,216 B
    __bf16* q_bf    = (__bf16*)(ws + 25165824);              // [B,H,S,D]
    __bf16* k_bf    = (__bf16*)(ws + 41943040);              // [B,H,S,D]
    __bf16* v_bf    = (__bf16*)(ws + 58720256);              // [B,H,D,S]
    __bf16* o_bf    = (__bf16*)(ws + 75497472);              // [B,S,H*D]
    float*  outp    = (float*)d_out;

    cvt_bf16_kernel<<<(QKV_N * DIM / 4 + 255) / 256, 256, 0, stream>>>(wqkv, wqkv_bf, QKV_N * DIM);
    cvt_bf16_kernel<<<(DIM * DIM / 4 + 255) / 256, 256, 0, stream>>>(wout, wout_bf, DIM * DIM);
    rmsnorm_kernel<<<NTOK, 256, 0, stream>>>(x, scale, xn_bf);
    qkv_gemm_rope_kernel<<<dim3(QKV_N / 64, NTOK / 128), 256, 0, stream>>>(
        xn_bf, wqkv_bf, theta, q_bf, k_bf, v_bf);
    attention_kernel<<<BATCH * HEADS * (SEQ / 128), 256, 0, stream>>>(q_bf, k_bf, v_bf, o_bf);
    outproj_kernel<<<dim3(DIM / 64, NTOK / 128), 256, 0, stream>>>(o_bf, wout_bf, x, outp);
}
